// Transformer_85882166050856
// MI455X (gfx1250) — compile-verified, matrix operand order checked by reading, not run on hardware
//
#include <hip/hip_runtime.h>
#include <cstdint>
#include <cstddef>

// ---------------------------------------------------------------------------
// Types for CDNA5 WMMA (wave32): v_wmma_f32_16x16x32_bf16
// ---------------------------------------------------------------------------
typedef __bf16 bf16;
typedef bf16  v16bf __attribute__((ext_vector_type(16)));
typedef float v8f   __attribute__((ext_vector_type(8)));

// Model constants (match reference)
constexpr int kB = 4, kS = 512, kD = 1024, kH = 16, kDH = 64;
constexpr int kV = 32000, kL = 4, kFF = 4096;
constexpr int kM = kB * kS;                 // 2048 tokens (enc and dec)
constexpr long long kDD = (long long)kD * kD;
constexpr long long kDF = (long long)kD * kFF;

__device__ __forceinline__ bf16 f2bf(float f) {
  union { float f; uint32_t u; } c; c.f = f;
  uint32_t r = (c.u + 0x7FFFu + ((c.u >> 16) & 1u)) >> 16;   // RNE
  union { uint16_t s; bf16 b; } o; o.s = (uint16_t)r;
  return o.b;
}

// --------------------------- async-to-LDS probe ----------------------------
#if __has_builtin(__builtin_amdgcn_global_load_async_to_lds_b128)
#define HAS_ASYNC 1
#else
#define HAS_ASYNC 0
#endif

// Builtin wants NATIVE clang vectors (vector_size(16) int), not HIP_vector_type.
// AS1 = global ("__device__" in clang-HIP diagnostics), AS3 = LDS.
typedef int nat_int4 __attribute__((vector_size(16)));
typedef __attribute__((address_space(1))) nat_int4 as1_nat4;
typedef __attribute__((address_space(3))) nat_int4 as3_nat4;

__device__ __forceinline__ void wait_async() {
#if HAS_ASYNC
#if __has_builtin(__builtin_amdgcn_s_wait_asynccnt)
  __builtin_amdgcn_s_wait_asynccnt(0);
#else
  asm volatile("s_wait_asynccnt 0" ::: "memory");
#endif
#endif
}

__device__ __forceinline__ void copy16_g2l(const bf16* __restrict__ g,
                                           bf16* __restrict__ l) {
#if HAS_ASYNC
  __builtin_amdgcn_global_load_async_to_lds_b128((as1_nat4*)g, (as3_nat4*)l,
                                                 0, 0);
#else
  *(uint4*)l = *(const uint4*)g;
#endif
}

// ---------------------------------------------------------------------------
// WMMA GEMM: C[M][N] = scale * (A_bf16[M][K] x B) + bias
//   COLK==0: B given as Bt[N][K] (pre-transposed, rows contiguous in K)
//   COLK==1: B given row-major [K][N] (transposed into LDS while staging)
// Wave tile: (WMF*16) x (WNF*16); wave grid WVM x WVN; block tile
// BM=WVM*WMF*16 x BN=WVN*WNF*16; K-step 32; double-buffered LDS, one
// barrier per K-step. Batched via grid.z (zb=z/Hdiv, zh=z%Hdiv).
// ---------------------------------------------------------------------------
template <int LDP>
__device__ __forceinline__ v16bf load_frag(const bf16* base, int lane) {
  // 16-bit A/Bt fragment (ISA 7.12.2): lanes 0-15 row r, K {0..7,16..23};
  // lanes 16-31 row r, K {8..15,24..31}.
  const bf16* p = base + (lane & 15) * LDP + ((lane >> 4) & 1) * 8;
  union { uint4 q[2]; v16bf v; } u;
  u.q[0] = *(const uint4*)p;
  u.q[1] = *(const uint4*)(p + 16);
  return u.v;
}

template <int COLK, int WMF, int WNF, int WVM, int WVN>
__global__ __launch_bounds__(WVM* WVN * 32) void k_gemm(
    const bf16* __restrict__ A, const bf16* __restrict__ B,
    const float* __restrict__ bias, float* __restrict__ C, bf16* __restrict__ Cb,
    int K, int lda, int ldb, int ldc,
    long long aB, long long aH, long long bB, long long bH,
    long long cB, long long cH, int Hdiv, float scale)
{
  constexpr int T  = WVM * WVN * 32;
  constexpr int BM = WVM * WMF * 16, BN = WVN * WNF * 16;
  constexpr int BK = 32, LDP = BK + 8;   // 80B row pitch: conflict-free b128
  __shared__ __align__(16) bf16 lA[2][BM * LDP];
  __shared__ __align__(16) bf16 lB[2][BN * LDP];

  const int z = blockIdx.z, zb = z / Hdiv, zh = z % Hdiv;
  A += (long long)zb * aB + (long long)zh * aH;
  B += (long long)zb * bB + (long long)zh * bH;
  const long long co = (long long)zb * cB + (long long)zh * cH;

  const int m0 = blockIdx.y * BM, n0 = blockIdx.x * BN;
  const int t = threadIdx.x, lane = t & 31, wave = t >> 5;
  const int wm = (wave / WVN) * WMF * 16;
  const int wn = (wave % WVN) * WNF * 16;

  v8f acc[WMF][WNF];
#pragma unroll
  for (int i = 0; i < WMF; ++i)
#pragma unroll
    for (int j = 0; j < WNF; ++j) acc[i][j] = (v8f){};

  auto stage = [&](int buf, int k0) {
    {  // A tile BM x 32
      constexpr int CH = (BM * BK / 8) / T;
#pragma unroll
      for (int c = 0; c < CH; ++c) {
        const int idx = t + c * T;
        const int r = idx >> 2, col = (idx & 3) * 8;
        copy16_g2l(A + (long long)(m0 + r) * lda + k0 + col,
                   &lA[buf][r * LDP + col]);
      }
    }
    if (COLK == 0) {  // Bt[N][K] tile BN x 32
      constexpr int CH = (BN * BK / 8) / T;
#pragma unroll
      for (int c = 0; c < CH; ++c) {
        const int idx = t + c * T;
        const int r = idx >> 2, col = (idx & 3) * 8;
        copy16_g2l(B + (long long)(n0 + r) * ldb + k0 + col,
                   &lB[buf][r * LDP + col]);
      }
    } else {          // B[K][N] tile 32 x BN, transpose into lB[n][k]
      constexpr int CH = (BK * BN / 8) / T;
#pragma unroll
      for (int c = 0; c < CH; ++c) {
        const int idx = t + c * T;
        const int kr = idx / (BN / 8), nc = (idx % (BN / 8)) * 8;
        bf16 tmp[8];
        *(uint4*)tmp = *(const uint4*)(B + (long long)(k0 + kr) * ldb + n0 + nc);
#pragma unroll
        for (int j = 0; j < 8; ++j) lB[buf][(nc + j) * LDP + kr] = tmp[j];
      }
    }
  };

  stage(0, 0);
  int buf = 0;
  for (int k0 = 0; k0 < K; k0 += BK) {
    wait_async();          // no-op on sync path
    __syncthreads();
    if (k0 + BK < K) stage(buf ^ 1, k0 + BK);
    v16bf fa[WMF], fb[WNF];
#pragma unroll
    for (int i = 0; i < WMF; ++i)
      fa[i] = load_frag<LDP>(&lA[buf][(wm + i * 16) * LDP], lane);
#pragma unroll
    for (int j = 0; j < WNF; ++j)
      fb[j] = load_frag<LDP>(&lB[buf][(wn + j * 16) * LDP], lane);
#pragma unroll
    for (int i = 0; i < WMF; ++i)
#pragma unroll
      for (int j = 0; j < WNF; ++j)
        acc[i][j] = __builtin_amdgcn_wmma_f32_16x16x32_bf16(
            false, fa[i], false, fb[j], (short)0, acc[i][j], false, false);
    buf ^= 1;
  }

  // C/D layout: VGPR k -> row (k | k+8), col = lane&15 (ISA 7.12.2)
  const int r0 = m0 + wm + ((lane >> 4) ? 8 : 0);
  const int c0 = n0 + wn + (lane & 15);
#pragma unroll
  for (int i = 0; i < WMF; ++i)
#pragma unroll
    for (int j = 0; j < WNF; ++j) {
      const int col = c0 + j * 16;
      const float bv = bias ? bias[col] : 0.f;
#pragma unroll
      for (int k = 0; k < 8; ++k) {
        const float v = acc[i][j][k] * scale + bv;
        const long long o = co + (long long)(r0 + i * 16 + k) * ldc + col;
        if (C)  C[o]  = v;
        if (Cb) Cb[o] = f2bf(v);
      }
    }
}

// ---------------------------------------------------------------------------
// Elementwise / reduction kernels (fp32 math, bf16 side-outputs for WMMA)
// ---------------------------------------------------------------------------
__global__ __launch_bounds__(256) void k_embed(
    const int* __restrict__ ids, const float* __restrict__ tok,
    const float* __restrict__ pos, float* __restrict__ x, int S)
{
  const long long row = blockIdx.x;
  const int id = ids[row];
  const int s  = (int)(row % S);
  const float* tr = tok + (long long)id * kD;
  const float* pr = pos + (long long)s * kD;
  float* xr = x + row * (long long)kD;
  for (int d = threadIdx.x; d < kD; d += 256) xr[d] = tr[d] + pr[d];
}

__global__ __launch_bounds__(256) void k_ln(
    const float* __restrict__ x, const float* __restrict__ g,
    const float* __restrict__ bt, float* __restrict__ y, bf16* __restrict__ yb)
{
  __shared__ float sh[8];
  const long long row = blockIdx.x;
  const float* xr = x + row * (long long)kD;
  const int t = threadIdx.x;
  float v[4], s = 0.f;
#pragma unroll
  for (int i = 0; i < 4; ++i) { v[i] = xr[t + i * 256]; s += v[i]; }
#pragma unroll
  for (int o = 16; o; o >>= 1) s += __shfl_xor(s, o, 32);
  if (!(t & 31)) sh[t >> 5] = s;
  __syncthreads();
  float mean = 0.f;
#pragma unroll
  for (int i = 0; i < 8; ++i) mean += sh[i];
  mean *= (1.f / kD);
  __syncthreads();
  float vs = 0.f;
#pragma unroll
  for (int i = 0; i < 4; ++i) { const float d = v[i] - mean; vs += d * d; }
#pragma unroll
  for (int o = 16; o; o >>= 1) vs += __shfl_xor(vs, o, 32);
  if (!(t & 31)) sh[t >> 5] = vs;
  __syncthreads();
  float var = 0.f;
#pragma unroll
  for (int i = 0; i < 8; ++i) var += sh[i];
  var *= (1.f / kD);
  const float rstd = rsqrtf(var + 1e-5f);
#pragma unroll
  for (int i = 0; i < 4; ++i) {
    const int c = t + i * 256;
    const float o = (v[i] - mean) * rstd * g[c] + bt[c];
    y[row * (long long)kD + c] = o;
    if (yb) yb[row * (long long)kD + c] = f2bf(o);
  }
}

// Masked softmax over Sk=512 keys; writes bf16 probs. grid.x = B*H*Sq.
__global__ __launch_bounds__(256) void k_softmax(
    const float* __restrict__ lg, bf16* __restrict__ pb,
    const int* __restrict__ ids, int Sk, int Sq, int causal)
{
  __shared__ float sh[8];
  const long long row = blockIdx.x;
  const int q = (int)(row % Sq);
  const int b = (int)(row / ((long long)Sq * kH));
  const float* lr = lg + row * Sk;
  bf16* pr = pb + row * Sk;
  const int* idr = ids + (long long)b * Sk;
  const int t = threadIdx.x;
  float v[2], mx = -3.4e38f;
#pragma unroll
  for (int i = 0; i < 2; ++i) {
    const int k = t + i * 256;
    float x = lr[k];
    if (causal && k > q) x += -1.0e9f;
    if (idr[k] == 0)     x += -1.0e9f;   // PAD mask
    v[i] = x;
    mx = fmaxf(mx, x);
  }
#pragma unroll
  for (int o = 16; o; o >>= 1) mx = fmaxf(mx, __shfl_xor(mx, o, 32));
  if (!(t & 31)) sh[t >> 5] = mx;
  __syncthreads();
  float m = sh[0];
#pragma unroll
  for (int i = 1; i < 8; ++i) m = fmaxf(m, sh[i]);
  __syncthreads();
  float s = 0.f;
#pragma unroll
  for (int i = 0; i < 2; ++i) { v[i] = __expf(v[i] - m); s += v[i]; }
#pragma unroll
  for (int o = 16; o; o >>= 1) s += __shfl_xor(s, o, 32);
  if (!(t & 31)) sh[t >> 5] = s;
  __syncthreads();
  float tot = 0.f;
#pragma unroll
  for (int i = 0; i < 8; ++i) tot += sh[i];
  const float inv = 1.f / tot;
#pragma unroll
  for (int i = 0; i < 2; ++i) pr[t + i * 256] = f2bf(v[i] * inv);
}

__global__ void k_add(const float* __restrict__ a, const float* __restrict__ b,
                      float* __restrict__ c, long long n) {
  const long long i = (long long)blockIdx.x * blockDim.x + threadIdx.x;
  if (i < n) c[i] = a[i] + b[i];
}

__global__ void k_cvt(const float* __restrict__ a, bf16* __restrict__ b, long long n) {
  const long long i = (long long)blockIdx.x * blockDim.x + threadIdx.x;
  if (i < n) b[i] = f2bf(a[i]);
}

__global__ void k_gelu(const float* __restrict__ a, bf16* __restrict__ b, long long n) {
  const long long i = (long long)blockIdx.x * blockDim.x + threadIdx.x;
  if (i < n) {
    const float v = a[i];
    b[i] = f2bf(0.5f * v * (1.f + erff(v * 0.70710678118654752f)));
  }
}

// W fp32 [Kd][Nd] -> Wt bf16 [Nd][Kd] (32x32 tile transpose through LDS)
__global__ __launch_bounds__(256) void k_cvtT(
    const float* __restrict__ W, bf16* __restrict__ Wt, int Kd, int Nd)
{
  __shared__ float tile[32][33];
  const int n0 = blockIdx.x * 32, k0 = blockIdx.y * 32;
  const int tx = threadIdx.x & 31, ty = threadIdx.x >> 5;
  for (int i = ty; i < 32; i += 8)
    tile[i][tx] = W[(long long)(k0 + i) * Nd + n0 + tx];
  __syncthreads();
  for (int i = ty; i < 32; i += 8)
    Wt[(long long)(n0 + i) * Kd + k0 + tx] = f2bf(tile[tx][i]);
}

// ---------------------------------------------------------------------------
// Host orchestration
// ---------------------------------------------------------------------------
extern "C" void kernel_launch(void* const* d_in, const int* in_sizes, int n_in,
                              void* d_out, int out_size, void* d_ws, size_t ws_size,
                              hipStream_t stream)
{
  (void)in_sizes; (void)n_in; (void)out_size; (void)ws_size;

  // jax pytree flatten order: dict keys sorted; 'W*' < 'b*' < 'ln*' (ASCII)
  const int*   input_ids = (const int*)  d_in[0];
  const float* dWk1  = (const float*)d_in[1];
  const float* dWk2  = (const float*)d_in[2];
  const float* dWm1  = (const float*)d_in[3];
  const float* dWm2  = (const float*)d_in[4];
  const float* dWo1  = (const float*)d_in[5];
  const float* dWo2  = (const float*)d_in[6];
  const float* dWq1  = (const float*)d_in[7];
  const float* dWq2  = (const float*)d_in[8];
  const float* dWv1  = (const float*)d_in[9];
  const float* dWv2  = (const float*)d_in[10];
  const float* dbk1  = (const float*)d_in[11];
  const float* dbk2  = (const float*)d_in[12];
  const float* dbm1  = (const float*)d_in[13];
  const float* dbm2  = (const float*)d_in[14];
  const float* dbo1  = (const float*)d_in[15];
  const float* dbo2  = (const float*)d_in[16];
  const float* dbq1  = (const float*)d_in[17];
  const float* dbq2  = (const float*)d_in[18];
  const float* dbv1  = (const float*)d_in[19];
  const float* dbv2  = (const float*)d_in[20];
  const float* dln1b = (const float*)d_in[21];
  const float* dln1g = (const float*)d_in[22];
  const float* dln2b = (const float*)d_in[23];
  const float* dln2g = (const float*)d_in[24];
  const float* dln3b = (const float*)d_in[25];
  const float* dln3g = (const float*)d_in[26];
  const float* dec_pos = (const float*)d_in[27];
  const float* dec_tok = (const float*)d_in[28];
  const float* eWk   = (const float*)d_in[29];
  const float* eWm1  = (const float*)d_in[30];
  const float* eWm2  = (const float*)d_in[31];
  const float* eWo   = (const float*)d_in[32];
  const float* eWq   = (const float*)d_in[33];
  const float* eWv   = (const float*)d_in[34];
  const float* ebk   = (const float*)d_in[35];
  const float* ebm1  = (const float*)d_in[36];
  const float* ebm2  = (const float*)d_in[37];
  const float* ebo   = (const float*)d_in[38];
  const float* ebq   = (const float*)d_in[39];
  const float* ebv   = (const float*)d_in[40];
  const float* eln1b = (const float*)d_in[41];
  const float* eln1g = (const float*)d_in[42];
  const float* eln2b = (const float*)d_in[43];
  const float* eln2g = (const float*)d_in[44];
  const float* eln3b = (const float*)d_in[45];
  const float* eln3g = (const float*)d_in[46];
  const float* enc_pos = (const float*)d_in[47];
  const float* enc_tok = (const float*)d_in[48];
  const float* head_W  = (const float*)d_in[49];
  const float* head_b  = (const float*)d_in[50];
  const int*   target_ids = (const int*)d_in[51];
  float* out = (float*)d_out;

  // Bump allocator over d_ws (deterministic, no hipMalloc)
  char* wp = (char*)d_ws;
  auto alloc  = [&](size_t bytes) -> void* {
    void* r = (void*)wp; wp += (bytes + 255) & ~(size_t)255; return r;
  };
  auto f32buf = [&](size_t n) { return (float*)alloc(n * sizeof(float)); };
  auto bfbuf  = [&](size_t n) { return (bf16*)alloc(n * sizeof(bf16)); };

  // bf16 transposed weights
  bf16 *eWqT = bfbuf(kL * kDD), *eWkT = bfbuf(kL * kDD), *eWvT = bfbuf(kL * kDD);
  bf16 *eWoT = bfbuf(kL * kDD), *eWm1T = bfbuf(kL * kDF), *eWm2T = bfbuf(kL * kDF);
  bf16 *dWq1T = bfbuf(kL * kDD), *dWk1T = bfbuf(kL * kDD), *dWv1T = bfbuf(kL * kDD);
  bf16 *dWo1T = bfbuf(kL * kDD), *dWq2T = bfbuf(kL * kDD), *dWk2T = bfbuf(kL * kDD);
  bf16 *dWv2T = bfbuf(kL * kDD), *dWo2T = bfbuf(kL * kDD);
  bf16 *dWm1T = bfbuf(kL * kDF), *dWm2T = bfbuf(kL * kDF);
  bf16 *headT = bfbuf((size_t)kV * kD);

  // Activations
  const size_t MD = (size_t)kM * kD, MF = (size_t)kM * kFF;
  const size_t LG = (size_t)kB * kH * kS * kS;
  float* xe  = f32buf(MD);  bf16* xeb = bfbuf(MD);
  float* yd  = f32buf(MD);  bf16* ydb = bfbuf(MD);
  float* xn  = f32buf(MD);  bf16* xnb = bfbuf(MD);
  bf16 *Qb = bfbuf(MD), *Kb = bfbuf(MD), *Vb = bfbuf(MD), *AOb = bfbuf(MD);
  float* Lg  = f32buf(LG);  bf16* Pb  = bfbuf(LG);
  float* ATf = f32buf(MD);
  float* T1  = f32buf(MD);
  float* X2  = f32buf(MD);
  float* T2  = f32buf(MD);  bf16* T2b  = bfbuf(MD);
  float* HDN = f32buf(MD);  bf16* HDNb = bfbuf(MD);
  float* MM  = f32buf(MF);  bf16* Gb   = bfbuf(MF);

  // Launch helpers
  auto cvtT = [&](const float* W, bf16* Wt, int Kd, int Nd) {
    dim3 g(Nd / 32, Kd / 32);
    k_cvtT<<<g, 256, 0, stream>>>(W, Wt, Kd, Nd);
  };
  // big config: block 128x128, wave 32x64 (2x4 frags -> 8 wmma / K-step)
  auto gemm = [&](const bf16* A, const bf16* Bt, const float* bias, float* C,
                  bf16* Cb, int M, int N, int K, int lda, int ldb, int ldc,
                  float scale) {
    dim3 g(N / 128, M / 128, 1);
    k_gemm<0, 2, 4, 4, 2><<<g, 256, 0, stream>>>(A, Bt, bias, C, Cb, K,
        lda, ldb, ldc, 0, 0, 0, 0, 0, 0, 1, scale);
  };
  auto gemm_qk = [&](const bf16* Q, const bf16* Kk, float* Cl) {
    dim3 g(kS / 128, kS / 128, kB * kH);   // per (b,h): [Sq x Sk] = Q Kt
    k_gemm<0, 2, 4, 4, 2><<<g, 256, 0, stream>>>(Q, Kk, nullptr, Cl, nullptr,
        kDH, kD, kD, kS,
        (long long)kS * kD, kDH,            // A strides (b, h)
        (long long)kS * kD, kDH,            // B strides
        (long long)kH * kS * kS, (long long)kS * kS,  // C strides
        kH, 0.125f);                        // 1/sqrt(64)
  };
  auto gemm_pv = [&](const bf16* P, const bf16* Vv, bf16* Cbb) {
    dim3 g(1, kS / 64, kB * kH);           // per (b,h): [Sq x 64] = P V
    k_gemm<1, 1, 2, 4, 2><<<g, 256, 0, stream>>>(P, Vv, nullptr, nullptr, Cbb,
        kS, kS, kD, kD,
        (long long)kH * kS * kS, (long long)kS * kS,  // A (probs)
        (long long)kS * kD, kDH,                      // B (V row-major)
        (long long)kS * kD, kDH,                      // C -> [b,q,h,d]
        kH, 1.f);
  };
  auto ln = [&](const float* x, const float* g, const float* b, float* y,
                bf16* yb) {
    k_ln<<<kM, 256, 0, stream>>>(x, g, b, y, yb);
  };
  auto softmax = [&](const int* key_ids, int causal) {
    k_softmax<<<kB * kH * kS, 256, 0, stream>>>(Lg, Pb, key_ids, kS, kS, causal);
  };
  auto add = [&](const float* a, const float* b, float* c, long long n) {
    k_add<<<(unsigned)((n + 255) / 256), 256, 0, stream>>>(a, b, c, n);
  };
  auto cvt = [&](const float* a, bf16* b, long long n) {
    k_cvt<<<(unsigned)((n + 255) / 256), 256, 0, stream>>>(a, b, n);
  };
  auto gelu = [&](const float* a, bf16* b, long long n) {
    k_gelu<<<(unsigned)((n + 255) / 256), 256, 0, stream>>>(a, b, n);
  };

  // ---- weight conversion (fp32 [K][N] -> bf16 [N][K]) ----
  for (int i = 0; i < kL; ++i) {
    cvtT(eWq + i * kDD, eWqT + i * kDD, kD, kD);
    cvtT(eWk + i * kDD, eWkT + i * kDD, kD, kD);
    cvtT(eWv + i * kDD, eWvT + i * kDD, kD, kD);
    cvtT(eWo + i * kDD, eWoT + i * kDD, kD, kD);
    cvtT(eWm1 + i * kDF, eWm1T + i * kDF, kD, kFF);
    cvtT(eWm2 + i * kDF, eWm2T + i * kDF, kFF, kD);
    cvtT(dWq1 + i * kDD, dWq1T + i * kDD, kD, kD);
    cvtT(dWk1 + i * kDD, dWk1T + i * kDD, kD, kD);
    cvtT(dWv1 + i * kDD, dWv1T + i * kDD, kD, kD);
    cvtT(dWo1 + i * kDD, dWo1T + i * kDD, kD, kD);
    cvtT(dWq2 + i * kDD, dWq2T + i * kDD, kD, kD);
    cvtT(dWk2 + i * kDD, dWk2T + i * kDD, kD, kD);
    cvtT(dWv2 + i * kDD, dWv2T + i * kDD, kD, kD);
    cvtT(dWo2 + i * kDD, dWo2T + i * kDD, kD, kD);
    cvtT(dWm1 + i * kDF, dWm1T + i * kDF, kD, kFF);
    cvtT(dWm2 + i * kDF, dWm2T + i * kDF, kFF, kD);
  }
  cvtT(head_W, headT, kD, kV);

  // ---- encoder ----
  k_embed<<<kM, 256, 0, stream>>>(input_ids, enc_tok, enc_pos, xe, kS);
  for (int i = 0; i < kL; ++i) {
    ln(xe, eln1g + i * kD, eln1b + i * kD, xn, xnb);
    gemm(xnb, eWqT + i * kDD, ebq + i * kD, nullptr, Qb, kM, kD, kD, kD, kD, kD, 1.f);
    gemm(xnb, eWkT + i * kDD, ebk + i * kD, nullptr, Kb, kM, kD, kD, kD, kD, kD, 1.f);
    gemm(xnb, eWvT + i * kDD, ebv + i * kD, nullptr, Vb, kM, kD, kD, kD, kD, kD, 1.f);
    gemm_qk(Qb, Kb, Lg);
    softmax(input_ids, /*causal=*/0);
    gemm_pv(Pb, Vb, AOb);
    gemm(AOb, eWoT + i * kDD, ebo + i * kD, ATf, nullptr, kM, kD, kD, kD, kD, kD, 1.f);
    ln(ATf, eln2g + i * kD, eln2b + i * kD, T1, nullptr);
    add(xn, T1, X2, (long long)MD);                        // x2 = xn + ln2(attn)
    ln(X2, eln3g + i * kD, eln3b + i * kD, T2, T2b);
    gemm(T2b, eWm1T + i * kDF, ebm1 + i * kFF, MM, nullptr, kM, kFF, kD, kD, kD, kFF, 1.f);
    gelu(MM, Gb, (long long)MF);
    gemm(Gb, eWm2T + i * kDF, ebm2 + i * kD, T1, nullptr, kM, kD, kFF, kFF, kFF, kD, 1.f);
    add(X2, T1, xe, (long long)MD);                        // x = x2 + mlp
  }
  cvt(xe, xeb, (long long)MD);                             // encoder output (bf16)

  // ---- decoder ----
  k_embed<<<kM, 256, 0, stream>>>(target_ids, dec_tok, dec_pos, yd, kS);
  for (int i = 0; i < kL; ++i) {
    ln(yd, dln1g + i * kD, dln1b + i * kD, xn, xnb);       // yn
    gemm(xnb, dWq1T + i * kDD, dbq1 + i * kD, nullptr, Qb, kM, kD, kD, kD, kD, kD, 1.f);
    gemm(xnb, dWk1T + i * kDD, dbk1 + i * kD, nullptr, Kb, kM, kD, kD, kD, kD, kD, 1.f);
    gemm(xnb, dWv1T + i * kDD, dbv1 + i * kD, nullptr, Vb, kM, kD, kD, kD, kD, kD, 1.f);
    gemm_qk(Qb, Kb, Lg);
    softmax(target_ids, /*causal=*/1);
    gemm_pv(Pb, Vb, AOb);
    gemm(AOb, dWo1T + i * kDD, dbo1 + i * kD, ATf, nullptr, kM, kD, kD, kD, kD, kD, 1.f);
    add(xn, ATf, T1, (long long)MD);                       // yn + sa
    ln(T1, dln2g + i * kD, dln2b + i * kD, HDN, HDNb);     // hdn
    // cross-attention
    gemm(HDNb, dWq2T + i * kDD, dbq2 + i * kD, nullptr, Qb, kM, kD, kD, kD, kD, kD, 1.f);
    gemm(xeb,  dWk2T + i * kDD, dbk2 + i * kD, nullptr, Kb, kM, kD, kD, kD, kD, kD, 1.f);
    gemm(xeb,  dWv2T + i * kDD, dbv2 + i * kD, nullptr, Vb, kM, kD, kD, kD, kD, kD, 1.f);
    gemm_qk(Qb, Kb, Lg);
    softmax(input_ids, /*causal=*/0);
    gemm_pv(Pb, Vb, AOb);
    gemm(AOb, dWo2T + i * kDD, dbo2 + i * kD, ATf, nullptr, kM, kD, kD, kD, kD, kD, 1.f);
    add(ATf, HDN, X2, (long long)MD);                      // x = ca + hdn
    ln(X2, dln3g + i * kD, dln3b + i * kD, T2, T2b);
    gemm(T2b, dWm1T + i * kDF, dbm1 + i * kFF, MM, nullptr, kM, kFF, kD, kD, kD, kFF, 1.f);
    gelu(MM, Gb, (long long)MF);
    gemm(Gb, dWm2T + i * kDF, dbm2 + i * kD, T1, nullptr, kM, kD, kFF, kFF, kFF, kD, 1.f);
    add(T1, X2, yd, (long long)MD);                        // y = mlp + x
  }

  // ---- output head: [2048 x 32000] ----
  cvt(yd, ydb, (long long)MD);
  gemm(ydb, headT, head_b, out, nullptr, kM, kV, kD, kD, kD, kV, 1.f);
}
